// GATv2_42992622633738
// MI455X (gfx1250) — matrix operand order repaired and checked by reading, not compiled, not run on hardware
//
#include <hip/hip_runtime.h>
#include <hip/hip_bf16.h>

// ---------------------------------------------------------------------------
// Two-layer GATv2 for MI455X (gfx1250, wave32).
//  - Node projections via v_wmma_f32_16x16x32_f16. A-tiles staged into LDS in
//    fragment order (ds_load_b128 in the hot loop); B weights pre-packed into
//    fragment order in scratch (contiguous global_load_b128 per lane).
//  - Edge softmax/aggregation via 3-pass segment ops with L2-resident atomics
//    (fs/fd/agg arrays fit in the 192MB L2); gathers vectorized to float4.
// ---------------------------------------------------------------------------

typedef __attribute__((ext_vector_type(16))) _Float16 v16h;
typedef __attribute__((ext_vector_type(8)))  float    v8f;

#define GAT_IN_DIM   128
#define GAT_HEADS    8
#define GAT_HID      32
#define GAT_F1       (GAT_HEADS * GAT_HID)   // 256
#define GAT_OUT      64
#define GAT_SLOPE    0.2f

// ---------------- helpers: order-preserving float<->uint for atomicMax ------
__device__ __forceinline__ unsigned f2ord(float x) {
    unsigned u = __float_as_uint(x);
    return (u & 0x80000000u) ? ~u : (u | 0x80000000u);
}
__device__ __forceinline__ float ord2f(unsigned u) {
    unsigned b = (u & 0x80000000u) ? (u ^ 0x80000000u) : ~u;
    return __uint_as_float(b);
}

// ---------------- weight pack: fp32 [K,N] -> fp16 B-fragment order ----------
// Fragment (nt, ks): 512 halves; lane L (m-half = L>>4, n = L&15), element e
// maps to W[(ks*32 + 8*(L>>4) + (e&7) + ((e&8)?16:0)) * Nout + nt*16 + (L&15)]
// (ISA 7.12.2 table order: lanes 0-15 hold K=0..7,16..23; lanes 16-31 hold K=8..15,24..31).
__global__ void pack_weight_kernel(const float* __restrict__ W,
                                   _Float16* __restrict__ Wp,
                                   int Kdim, int Nout) {
    const int ksteps = Kdim >> 5;
    const int ntiles = Nout >> 4;
    const int total  = ntiles * ksteps * 512;
    int i = blockIdx.x * blockDim.x + threadIdx.x;
    if (i >= total) return;
    int e    = i & 15;
    int lane = (i >> 4) & 31;
    int frag = i >> 9;                 // nt*ksteps + ks
    int ks   = frag % ksteps;
    int nt   = frag / ksteps;
    int k    = ks * 32 + (lane >> 4) * 8 + (e & 7) + ((e & 8) ? 16 : 0);
    int n    = nt * 16 + (lane & 15);
    Wp[i] = (_Float16)W[(size_t)k * Nout + n];
}

// ---------------- dual GEMM: Ya = X*Wa, Yb = X*Wb  (WMMA f16->f32) ----------
// One 16-row M-tile per block (8 waves). LDS holds the A tile pre-swizzled to
// fragment order, so the hot loop is 2x ds_load_b128 + 2x global_load_b128 +
// one v_wmma per k-step per (weight,ntile) combo.
__global__ void gemm_dual_wmma_kernel(const float* __restrict__ X, int M,
                                      int Kdim, int Nout,
                                      const _Float16* __restrict__ Wpa,
                                      const _Float16* __restrict__ Wpb,
                                      float* __restrict__ Ya,
                                      float* __restrict__ Yb) {
    __shared__ _Float16 sA[16 * 256] __attribute__((aligned(32)));  // 8 KB
    const int mtile = blockIdx.x;
    const int m0    = mtile * 16;

    // Stage & swizzle the 16 x Kdim tile of X into A-fragment order:
    //   element (m, k): ks=k>>5, kk=k&31, half=(kk>>3)&1,
    //                   e=(kk&7)|((kk&16)?8:0), lane=(half<<4)|m
    for (int idx = threadIdx.x; idx < 16 * Kdim; idx += blockDim.x) {
        int m  = idx / Kdim, k = idx - m * Kdim;
        int ks = k >> 5, kk = k & 31;
        int half = (kk >> 3) & 1;
        int e    = (kk & 7) | ((kk & 16) ? 8 : 0);
        int lane = (half << 4) | m;
        float v = (m0 + m < M) ? X[(size_t)(m0 + m) * Kdim + k] : 0.0f;
        sA[ks * 512 + lane * 16 + e] = (_Float16)v;
    }
    __syncthreads();

    const int lane   = threadIdx.x & 31;
    const int wave   = threadIdx.x >> 5;
    const int nwv    = blockDim.x >> 5;
    const int half   = lane >> 4;
    const int lanen  = lane & 15;
    const int ntiles = Nout >> 4;
    const int ksteps = Kdim >> 5;
    const bool full  = (m0 + 16 <= M);

    for (int combo = wave; combo < 2 * ntiles; combo += nwv) {
        const int sel = combo & 1;
        const int nt  = combo >> 1;
        const _Float16* __restrict__ Wp = sel ? Wpb : Wpa;
        float* __restrict__ Y           = sel ? Yb : Ya;

        v8f c = {};
        for (int ks = 0; ks < ksteps; ++ks) {
            v16h a = *(const v16h*)(&sA[ks * 512 + lane * 16]);
            v16h b = *(const v16h*)(Wp + ((size_t)(nt * ksteps + ks)) * 512
                                       + lane * 16);
            c = __builtin_amdgcn_wmma_f32_16x16x32_f16(
                    false, a, false, b, (short)0, c, false, false);
        }

        const size_t col = (size_t)nt * 16 + lanen;
        if (full) {
            #pragma unroll
            for (int r = 0; r < 8; ++r)
                Y[(size_t)(m0 + r + half * 8) * Nout + col] = c[r];
        } else {
            #pragma unroll
            for (int r = 0; r < 8; ++r) {
                int m = m0 + r + half * 8;
                if (m < M) Y[(size_t)m * Nout + col] = c[r];
            }
        }
    }
}

// ---------------- edge pass 1: logits + segment max -------------------------
// thread i = (edge, head). logit = sum_d attn[h,d]*leakyrelu(fs[src]+fd[dst])
__global__ void edge_logits_kernel(const float* __restrict__ fs,
                                   const float* __restrict__ fd,
                                   const int* __restrict__ src,
                                   const int* __restrict__ dst,
                                   const float* __restrict__ attn,
                                   float* __restrict__ logits,
                                   unsigned* __restrict__ menc,
                                   int E, int H, int D) {
    int i = blockIdx.x * blockDim.x + threadIdx.x;
    if (i >= E * H) return;
    int e = i / H, h = i - e * H;
    int F = H * D;
    const float4* __restrict__ ps =
        (const float4*)(fs + (size_t)src[e] * F + h * D);
    const float4* __restrict__ pd =
        (const float4*)(fd + (size_t)dst[e] * F + h * D);
    const float4* __restrict__ pa = (const float4*)(attn + h * D);
    float acc = 0.0f;
    #pragma unroll 4
    for (int j = 0; j < (D >> 2); ++j) {
        float4 a = ps[j], b = pd[j], w = pa[j];
        float v;
        v = a.x + b.x; v = (v > 0.0f) ? v : GAT_SLOPE * v; acc = fmaf(v, w.x, acc);
        v = a.y + b.y; v = (v > 0.0f) ? v : GAT_SLOPE * v; acc = fmaf(v, w.y, acc);
        v = a.z + b.z; v = (v > 0.0f) ? v : GAT_SLOPE * v; acc = fmaf(v, w.z, acc);
        v = a.w + b.w; v = (v > 0.0f) ? v : GAT_SLOPE * v; acc = fmaf(v, w.w, acc);
    }
    logits[i] = acc;
    atomicMax(&menc[(size_t)dst[e] * H + h], f2ord(acc));
}

// ---------------- edge pass 2: exp + denom + unnormalized aggregate ---------
__global__ void edge_scatter_kernel(const float* __restrict__ fs,
                                    const int* __restrict__ src,
                                    const int* __restrict__ dst,
                                    const float* __restrict__ logits,
                                    const unsigned* __restrict__ menc,
                                    float* __restrict__ denom,
                                    float* __restrict__ agg,
                                    int E, int H, int D) {
    int i = blockIdx.x * blockDim.x + threadIdx.x;
    if (i >= E * H) return;
    int e = i / H, h = i - e * H;
    int F = H * D;
    int d = dst[e];
    float m  = ord2f(menc[(size_t)d * H + h]);
    float ex = __expf(logits[i] - m);
    atomicAdd(&denom[(size_t)d * H + h], ex);
    const float4* __restrict__ ps =
        (const float4*)(fs + (size_t)src[e] * F + h * D);
    float* __restrict__ pa = agg + (size_t)d * F + h * D;
    #pragma unroll 4
    for (int j = 0; j < (D >> 2); ++j) {
        float4 a = ps[j];
        atomicAdd(&pa[4 * j + 0], ex * a.x);
        atomicAdd(&pa[4 * j + 1], ex * a.y);
        atomicAdd(&pa[4 * j + 2], ex * a.z);
        atomicAdd(&pa[4 * j + 3], ex * a.w);
    }
}

// ---------------- per-node finalize: normalize + bias (+ optional ELU) ------
__global__ void finalize_kernel(const float* __restrict__ agg,
                                const float* __restrict__ denom,
                                const float* __restrict__ bias,
                                float* __restrict__ out,
                                int Nn, int H, int D, int do_elu) {
    int i = blockIdx.x * blockDim.x + threadIdx.x;
    int F = H * D;
    if (i >= Nn * F) return;
    int n = i / F, f = i - n * F;
    int h = f / D;
    float dn = denom[(size_t)n * H + h];
    float v  = (dn > 0.0f) ? agg[i] / dn : 0.0f;
    v += bias[f];
    if (do_elu) v = (v > 0.0f) ? v : (__expf(v) - 1.0f);
    out[i] = v;
}

// ---------------------------------------------------------------------------
extern "C" void kernel_launch(void* const* d_in, const int* in_sizes, int n_in,
                              void* d_out, int out_size, void* d_ws, size_t ws_size,
                              hipStream_t stream) {
    const float* h      = (const float*)d_in[0];
    const int*   src    = (const int*)  d_in[1];
    const int*   dst    = (const int*)  d_in[2];
    const float* W1s    = (const float*)d_in[3];
    const float* W1d    = (const float*)d_in[4];
    const float* attn1  = (const float*)d_in[5];
    const float* b1     = (const float*)d_in[6];
    const float* W2s    = (const float*)d_in[7];
    const float* W2d    = (const float*)d_in[8];
    const float* attn2  = (const float*)d_in[9];
    const float* b2     = (const float*)d_in[10];
    float*       out    = (float*)d_out;

    const int N = in_sizes[0] / GAT_IN_DIM;
    const int E = in_sizes[1];

    // ---- deterministic workspace carve-out (256B aligned) ----
    uintptr_t base = (uintptr_t)d_ws;
    size_t off = 0;
    auto carve = [&](size_t bytes) -> uintptr_t {
        uintptr_t p = base + off;
        off = (off + bytes + 255) & ~(size_t)255;
        return p;
    };
    _Float16* w1s_p  = (_Float16*)carve((size_t)GAT_IN_DIM * GAT_F1 * 2);
    _Float16* w1d_p  = (_Float16*)carve((size_t)GAT_IN_DIM * GAT_F1 * 2);
    _Float16* w2s_p  = (_Float16*)carve((size_t)GAT_F1 * GAT_OUT * 2);
    _Float16* w2d_p  = (_Float16*)carve((size_t)GAT_F1 * GAT_OUT * 2);
    float*    fs1    = (float*)   carve((size_t)N * GAT_F1 * 4);
    float*    fd1    = (float*)   carve((size_t)N * GAT_F1 * 4);
    float*    agg1   = (float*)   carve((size_t)N * GAT_F1 * 4);   // becomes h1
    float*    logit1 = (float*)   carve((size_t)E * GAT_HEADS * 4);
    unsigned* menc1  = (unsigned*)carve((size_t)N * GAT_HEADS * 4);
    float*    den1   = (float*)   carve((size_t)N * GAT_HEADS * 4);
    float*    fs2    = (float*)   carve((size_t)N * GAT_OUT * 4);
    float*    fd2    = (float*)   carve((size_t)N * GAT_OUT * 4);
    float*    agg2   = (float*)   carve((size_t)N * GAT_OUT * 4);
    float*    logit2 = (float*)   carve((size_t)E * 4);
    unsigned* menc2  = (unsigned*)carve((size_t)N * 4);
    float*    den2   = (float*)   carve((size_t)N * 4);
    (void)ws_size; (void)n_in; (void)out_size;

    // ---- pack weights to fp16 fragment order, once per call ----
    {
        int n1 = GAT_IN_DIM * GAT_F1, n2 = GAT_F1 * GAT_OUT;
        pack_weight_kernel<<<(n1 + 255) / 256, 256, 0, stream>>>(
            W1s, w1s_p, GAT_IN_DIM, GAT_F1);
        pack_weight_kernel<<<(n1 + 255) / 256, 256, 0, stream>>>(
            W1d, w1d_p, GAT_IN_DIM, GAT_F1);
        pack_weight_kernel<<<(n2 + 255) / 256, 256, 0, stream>>>(
            W2s, w2s_p, GAT_F1, GAT_OUT);
        pack_weight_kernel<<<(n2 + 255) / 256, 256, 0, stream>>>(
            W2d, w2d_p, GAT_F1, GAT_OUT);
    }

    const int mtiles = (N + 15) / 16;

    // ================= Layer 1 =================
    gemm_dual_wmma_kernel<<<mtiles, 256, 0, stream>>>(
        h, N, GAT_IN_DIM, GAT_F1, w1s_p, w1d_p, fs1, fd1);

    hipMemsetAsync(menc1, 0, (size_t)N * GAT_HEADS * 4, stream);  // ordered -inf
    hipMemsetAsync(den1,  0, (size_t)N * GAT_HEADS * 4, stream);
    hipMemsetAsync(agg1,  0, (size_t)N * GAT_F1 * 4,    stream);

    {
        int work = E * GAT_HEADS;
        edge_logits_kernel<<<(work + 255) / 256, 256, 0, stream>>>(
            fs1, fd1, src, dst, attn1, logit1, menc1, E, GAT_HEADS, GAT_HID);
        edge_scatter_kernel<<<(work + 255) / 256, 256, 0, stream>>>(
            fs1, src, dst, logit1, menc1, den1, agg1, E, GAT_HEADS, GAT_HID);
        int nf = N * GAT_F1;  // normalize + bias + ELU, in place -> h1
        finalize_kernel<<<(nf + 255) / 256, 256, 0, stream>>>(
            agg1, den1, b1, agg1, N, GAT_HEADS, GAT_HID, /*elu=*/1);
    }

    // ================= Layer 2 =================
    gemm_dual_wmma_kernel<<<mtiles, 256, 0, stream>>>(
        agg1 /* = h1 */, N, GAT_F1, GAT_OUT, w2s_p, w2d_p, fs2, fd2);

    hipMemsetAsync(menc2, 0, (size_t)N * 4, stream);
    hipMemsetAsync(den2,  0, (size_t)N * 4, stream);
    hipMemsetAsync(agg2,  0, (size_t)N * GAT_OUT * 4, stream);

    {
        int work = E;  // H = 1
        edge_logits_kernel<<<(work + 255) / 256, 256, 0, stream>>>(
            fs2, fd2, src, dst, attn2, logit2, menc2, E, 1, GAT_OUT);
        edge_scatter_kernel<<<(work + 255) / 256, 256, 0, stream>>>(
            fs2, src, dst, logit2, menc2, den2, agg2, E, 1, GAT_OUT);
        int nf = N * GAT_OUT;  // normalize + bias; mean over 1 head = identity
        finalize_kernel<<<(nf + 255) / 256, 256, 0, stream>>>(
            agg2, den2, b2, out, N, 1, GAT_OUT, /*elu=*/0);
    }
}
